// GCN_Fusion_85916525789536
// MI455X (gfx1250) — matrix operand order repaired and checked by reading, not compiled, not run on hardware
//
#include <hip/hip_runtime.h>

typedef __attribute__((ext_vector_type(16))) __bf16 v16bf;
typedef __attribute__((ext_vector_type(8)))  __bf16 v8bf;
typedef __attribute__((ext_vector_type(8)))  float  v8f;

constexpr int Bn   = 2;
constexpr int Cc   = 64;
constexpr int Nn   = 2304;     // 48*48
constexpr int NT   = Nn / 16;  // 144 tiles
constexpr int NPAD = 4096;     // padded sort length (pow2 >= Nn)

// ---------------------------------------------------------------------------
// WMMA fragment builders (bf16 16x16x32, wave32), operands pre-staged in bf16.
// A (16x32, MxK) row-major: lane L -> M = L&15, h = L>>4.
//   frag = row[k0+8h .. +7] ++ row[k0+16+8h .. +7]  (two 16B chunks)
// B (32x16, KxN) with K-contiguous storage per column: lane -> N = L&15.
//   frag = col[k0+8h .. +7] ++ col[k0+16+8h .. +7]  (two 16B chunks)
// C/D (16x16 f32): lane -> N = L&15; acc[r] -> M = r + 8h
// ---------------------------------------------------------------------------
__device__ inline v16bf frag_a_bf16(const __bf16* __restrict__ row, int k0, int h) {
  v8bf lo = *(const v8bf*)(row + k0 + 8 * h);
  v8bf hi = *(const v8bf*)(row + k0 + 16 + 8 * h);
  v16bf a;
#pragma unroll
  for (int e = 0; e < 8; ++e) { a[e] = lo[e]; a[8 + e] = hi[e]; }
  return a;
}

__device__ inline v16bf frag_b_kcontig(const __bf16* __restrict__ colbase, int k0, int h) {
  v8bf lo = *(const v8bf*)(colbase + k0 + 8 * h);
  v8bf hi = *(const v8bf*)(colbase + k0 + 16 + 8 * h);
  v16bf b;
#pragma unroll
  for (int e = 0; e < 8; ++e) { b[e] = lo[e]; b[8 + e] = hi[e]; }
  return b;
}

// ---------------------------------------------------------------------------
// gram_kernel: G[i][j] = dot(Xrow_i, Yrow_j) over d=Cc, X/Y bf16 row-major.
// mode 0: euclidean self  -> sqrt(max(sq[i]+sq[j]-2g,0))
// mode 1: cosine          -> 1 - g        (X,Y row-normalized)
// mode 2: raw gram        -> g
// ---------------------------------------------------------------------------
__global__ __launch_bounds__(32)
void gram_kernel(const __bf16* __restrict__ X, const __bf16* __restrict__ Y,
                 float* __restrict__ out, const float* __restrict__ sq, int mode) {
  const int b = blockIdx.z, ti = blockIdx.x, tj = blockIdx.y;
  const int lane = threadIdx.x, h = lane >> 4, col = lane & 15;
  const __bf16* xr = X + ((size_t)b * Nn + ti * 16 + col) * Cc;
  const __bf16* yr = Y + ((size_t)b * Nn + tj * 16 + col) * Cc;
  v8f acc = {};
#pragma unroll
  for (int k0 = 0; k0 < Cc; k0 += 32) {
    v16bf a  = frag_a_bf16(xr, k0, h);      // A row i
    v16bf bm = frag_b_kcontig(yr, k0, h);   // B col j = Y row j (K-contig)
    acc = __builtin_amdgcn_wmma_f32_16x16x32_bf16(false, a, false, bm,
                                                  (short)0, acc, false, false);
  }
  const int j = tj * 16 + col;
  float* Ob = out + (size_t)b * Nn * Nn;
  const float* sqb = sq ? (sq + (size_t)b * Nn) : nullptr;
#pragma unroll
  for (int r = 0; r < 8; ++r) {
    int i = ti * 16 + r + 8 * h;
    float g = acc[r], val;
    if (mode == 0)      val = sqrtf(fmaxf(sqb[i] + sqb[j] - 2.0f * g, 0.0f));
    else if (mode == 1) val = 1.0f - g;
    else                val = g;
    Ob[(size_t)i * Nn + j] = val;
  }
}

// ---------------------------------------------------------------------------
// gemm_acc: out[b][ch][i] += sum_k P[b][i][k] * Xcm[b][ch][k]
// P bf16 row-major [n][n]; Xcm bf16 in ORIGINAL [b][C][n] layout (K-contig).
// ---------------------------------------------------------------------------
__global__ __launch_bounds__(32)
void gemm_acc_kernel(const __bf16* __restrict__ P, const __bf16* __restrict__ Xcm,
                     float* __restrict__ out) {
  const int b = blockIdx.z, ti = blockIdx.x, tc = blockIdx.y;
  const int lane = threadIdx.x, h = lane >> 4, col = lane & 15;
  const int ch = tc * 16 + col;
  const __bf16* pr = P + ((size_t)b * Nn + ti * 16 + col) * Nn;
  const __bf16* fc = Xcm + ((size_t)b * Cc + ch) * Nn;
  v8f acc = {};
#pragma unroll 4
  for (int k0 = 0; k0 < Nn; k0 += 32) {
    v16bf a  = frag_a_bf16(pr, k0, h);
    v16bf bm = frag_b_kcontig(fc, k0, h);
    acc = __builtin_amdgcn_wmma_f32_16x16x32_bf16(false, a, false, bm,
                                                  (short)0, acc, false, false);
  }
#pragma unroll
  for (int r = 0; r < 8; ++r) {
    int i = ti * 16 + r + 8 * h;
    size_t idx = ((size_t)b * Cc + ch) * Nn + i;
    out[idx] += acc[r];
  }
}

// ---------------------------------------------------------------------------
// cast x (f32 [b][C][n]) -> bf16 same layout (B operand of final GEMM)
// ---------------------------------------------------------------------------
__global__ void cast_kernel(const float* __restrict__ x, __bf16* __restrict__ xb) {
  int i = blockIdx.x * blockDim.x + threadIdx.x;
  if (i < Bn * Cc * Nn) xb[i] = (__bf16)x[i];
}

// prep: x [b][C][n] -> fTb bf16 [b][n][C], fnb bf16 (row-normalized), sq f32
__global__ void prep_kernel(const float* __restrict__ x, __bf16* __restrict__ fTb,
                            __bf16* __restrict__ fnb, float* __restrict__ sq) {
  int idx = blockIdx.x * blockDim.x + threadIdx.x;
  if (idx >= Bn * Nn) return;
  int b = idx / Nn, i = idx % Nn;
  const float* xb = x + (size_t)b * Cc * Nn;
  float row[Cc];
  float s = 0.f;
  for (int ch = 0; ch < Cc; ++ch) {
    float v = xb[(size_t)ch * Nn + i];
    row[ch] = v; s += v * v;
  }
  sq[idx] = s;
  float inv = 1.0f / fmaxf(sqrtf(s), 1e-12f);
  __bf16* frow = fTb + (size_t)idx * Cc;
  __bf16* nrow = fnb + (size_t)idx * Cc;
  for (int ch = 0; ch < Cc; ++ch) {
    frow[ch] = (__bf16)row[ch];
    nrow[ch] = (__bf16)(row[ch] * inv);
  }
}

// y[b][i][o] = bias[o] + sum_ch w[o][ch] * x[b][ch][i]  -> bf16 [b][n][C]
__global__ void conv1x1_kernel(const float* __restrict__ x, const float* __restrict__ w,
                               const float* __restrict__ bias, __bf16* __restrict__ y) {
  int idx = blockIdx.x * blockDim.x + threadIdx.x;
  if (idx >= Bn * Nn * Cc) return;
  int o = idx % Cc, i = (idx / Cc) % Nn, b = idx / (Cc * Nn);
  const float* xb = x + (size_t)b * Cc * Nn;
  float s = bias[o];
  for (int ch = 0; ch < Cc; ++ch) s += w[o * Cc + ch] * xb[(size_t)ch * Nn + i];
  y[idx] = (__bf16)s;
}

__global__ void zero_kernel(float* __restrict__ p, int nElem) {
  int i = blockIdx.x * blockDim.x + threadIdx.x;
  if (i < nElem) p[i] = 0.f;
}

// per-batch sum / sumsq over D
__global__ __launch_bounds__(256)
void stats_kernel(const float* __restrict__ D, float* __restrict__ stats) {
  int b = blockIdx.y;
  const float* Db = D + (size_t)b * Nn * Nn;
  float s = 0.f, ss = 0.f;
  for (size_t i = blockIdx.x * blockDim.x + threadIdx.x; i < (size_t)Nn * Nn;
       i += (size_t)gridDim.x * blockDim.x) {
    float v = Db[i]; s += v; ss += v * v;
  }
  __shared__ float r0[256], r1[256];
  r0[threadIdx.x] = s; r1[threadIdx.x] = ss; __syncthreads();
  for (int o = 128; o > 0; o >>= 1) {
    if (threadIdx.x < o) { r0[threadIdx.x] += r0[threadIdx.x + o];
                           r1[threadIdx.x] += r1[threadIdx.x + o]; }
    __syncthreads();
  }
  if (threadIdx.x == 0) {
    atomicAdd(&stats[b * 2 + 0], r0[0]);
    atomicAdd(&stats[b * 2 + 1], r1[0]);
  }
}

// ---------------------------------------------------------------------------
// adjacency: one block per (b,row). LDS bitonic sort of (D,idx) keys,
// chunked scan of sorted probs, cum>0.8 cutoff, T threshold, write 0/1 adj.
// ---------------------------------------------------------------------------
__global__ __launch_bounds__(256)
void adjacency_kernel(const float* __restrict__ D, const float* __restrict__ stats,
                      float* __restrict__ A) {
  const int row = blockIdx.x, b = blockIdx.y, tid = threadIdx.x;
  const float* Dr = D + ((size_t)b * Nn + row) * Nn;
  float* Ar = A + ((size_t)b * Nn + row) * Nn;

  __shared__ unsigned long long keys[NPAD];   // 32 KB
  __shared__ float cum[NPAD];                 // 16 KB
  __shared__ float chunk[256];
  __shared__ float red[256];
  __shared__ int firstOver;
  __shared__ float sh_dmax, sh_Z, sh_ent;

  // pass 1: row max, Z = sum exp(-D/tau)   (D>=0, tau=10 -> no overflow)
  float lmax = -1e30f, lz = 0.f;
  for (int j = tid; j < Nn; j += 256) {
    float v = Dr[j];
    lmax = fmaxf(lmax, v);
    lz += __expf(-v * 0.1f);
  }
  red[tid] = lmax; __syncthreads();
  for (int o = 128; o > 0; o >>= 1) { if (tid < o) red[tid] = fmaxf(red[tid], red[tid + o]); __syncthreads(); }
  if (tid == 0) sh_dmax = red[0];
  __syncthreads();
  red[tid] = lz; __syncthreads();
  for (int o = 128; o > 0; o >>= 1) { if (tid < o) red[tid] += red[tid + o]; __syncthreads(); }
  if (tid == 0) sh_Z = red[0];
  __syncthreads();
  const float Z = sh_Z;

  // pass 2: entropy
  float lent = 0.f;
  for (int j = tid; j < Nn; j += 256) {
    float p = __expf(-Dr[j] * 0.1f) / Z;
    lent -= p * __logf(p + 1e-8f);
  }
  red[tid] = lent; __syncthreads();
  for (int o = 128; o > 0; o >>= 1) { if (tid < o) red[tid] += red[tid + o]; __syncthreads(); }
  if (tid == 0) { sh_ent = red[0]; firstOver = NPAD; }

  // keys: (float bits of D, order-preserving for D>=0) << 32 | index
  __syncthreads();
  for (int j = tid; j < NPAD; j += 256) {
    if (j < Nn)
      keys[j] = ((unsigned long long)__float_as_uint(Dr[j]) << 32) | (unsigned int)j;
    else
      keys[j] = 0xFFFFFFFFFFFFFFFFull;
  }
  __syncthreads();

  // bitonic sort ascending
  for (int ksz = 2; ksz <= NPAD; ksz <<= 1)
    for (int jsz = ksz >> 1; jsz > 0; jsz >>= 1) {
      for (int i = tid; i < NPAD; i += 256) {
        int ixj = i ^ jsz;
        if (ixj > i) {
          bool up = ((i & ksz) == 0);
          unsigned long long a = keys[i], c = keys[ixj];
          if (up ? (a > c) : (a < c)) { keys[i] = c; keys[ixj] = a; }
        }
      }
      __syncthreads();
    }

  // sorted probs (prob is monotone in D: recover from key bits)
  for (int r = tid; r < NPAD; r += 256) {
    float p = 0.f;
    if (r < Nn) {
      float dv = __uint_as_float((unsigned int)(keys[r] >> 32));
      p = __expf(-dv * 0.1f) / Z;
    }
    cum[r] = p;
  }
  __syncthreads();

  // chunked inclusive scan (16 elems/thread)
  {
    int base = tid * 16; float s = 0.f;
#pragma unroll
    for (int e = 0; e < 16; ++e) { s += cum[base + e]; cum[base + e] = s; }
    chunk[tid] = s;
  }
  __syncthreads();
  if (tid == 0) {
    float acc = 0.f;
    for (int k = 0; k < 256; ++k) { float t = chunk[k]; chunk[k] = acc; acc += t; }
  }
  __syncthreads();
  {
    int base = tid * 16; float off = chunk[tid];
#pragma unroll
    for (int e = 0; e < 16; ++e) cum[base + e] += off;
  }
  __syncthreads();

  // first rank with cum > 0.8 (cum is nondecreasing)
  for (int r = tid; r < NPAD; r += 256)
    if (cum[r] > 0.8f) { atomicMin(&firstOver, r); break; }
  __syncthreads();

  int num = (firstOver < NPAD) ? firstOver : Nn;     // num_edges
  unsigned long long cutkey = keys[num];             // pads=MAX -> select all

  const float N2 = (float)Nn * (float)Nn;
  float mu = stats[b * 2 + 0] / N2;
  float var = (stats[b * 2 + 1] - N2 * mu * mu) / (N2 - 1.0f);
  float sigma = sqrtf(fmaxf(var, 0.f));
  float decay = __expf(-sh_ent);                     // ETA = 1.0
  float invdmax = 1.0f / sh_dmax;

  for (int j = tid; j < Nn; j += 256) {
    float dv = Dr[j];
    unsigned long long kj =
        ((unsigned long long)__float_as_uint(dv) << 32) | (unsigned int)j;
    float T = mu + 0.08f * sigma + 0.01f * (1.0f - dv * invdmax) * decay;
    Ar[j] = (kj < cutkey && dv < T) ? 1.0f : 0.0f;
  }
}

// P = softmax_row(A .* S), emitted directly in bf16 for the WMMA GEMM
__global__ __launch_bounds__(256)
void attn_kernel(const float* __restrict__ A, const float* __restrict__ S,
                 __bf16* __restrict__ P) {
  const int row = blockIdx.x, b = blockIdx.y, tid = threadIdx.x;
  const float* Ar = A + ((size_t)b * Nn + row) * Nn;
  const float* Sr = S + ((size_t)b * Nn + row) * Nn;
  __bf16* Pr = P + ((size_t)b * Nn + row) * Nn;
  __shared__ float vrow[Nn];
  __shared__ float red[256];
  float lmax = -1e30f;
  for (int j = tid; j < Nn; j += 256) {
    float v = Ar[j] * Sr[j];
    vrow[j] = v;
    lmax = fmaxf(lmax, v);
  }
  red[tid] = lmax; __syncthreads();
  for (int o = 128; o > 0; o >>= 1) { if (tid < o) red[tid] = fmaxf(red[tid], red[tid + o]); __syncthreads(); }
  float m = red[0];
  __syncthreads();
  float ls = 0.f;
  for (int j = tid; j < Nn; j += 256) { float e = __expf(vrow[j] - m); vrow[j] = e; ls += e; }
  red[tid] = ls; __syncthreads();
  for (int o = 128; o > 0; o >>= 1) { if (tid < o) red[tid] += red[tid + o]; __syncthreads(); }
  float inv = 1.0f / red[0];
  for (int j = tid; j < Nn; j += 256) Pr[j] = (__bf16)(vrow[j] * inv);
}

// ---------------------------------------------------------------------------
extern "C" void kernel_launch(void* const* d_in, const int* in_sizes, int n_in,
                              void* d_out, int out_size, void* d_ws, size_t ws_size,
                              hipStream_t stream) {
  (void)in_sizes; (void)n_in; (void)out_size; (void)ws_size;
  const float* x[3]    = {(const float*)d_in[0], (const float*)d_in[1], (const float*)d_in[2]};
  const float* w[3]    = {(const float*)d_in[3], (const float*)d_in[5], (const float*)d_in[7]};
  const float* bias[3] = {(const float*)d_in[4], (const float*)d_in[6], (const float*)d_in[8]};
  float* out = (float*)d_out;

  // ---- workspace carve-up: f32 region first, then bf16 region ----
  const size_t fsz = (size_t)Bn * Nn * Cc;   // per-tensor feature elements
  const size_t nsq = (size_t)Bn * Nn * Nn;   // n x n per-batch-pair elements
  float* wsf  = (float*)d_ws;
  float* Dbuf  = wsf;                        // [B][n][n] f32
  float* Abuf  = Dbuf + nsq;                 // [B][n][n] f32
  float* stats = Abuf + nsq;                 // 4 f32
  float* sqp   = stats + 4;                  // 3 * B*n f32
  float* sq[3];
  for (int s = 0; s < 3; ++s) sq[s] = sqp + (size_t)s * Bn * Nn;

  __bf16* wsb = (__bf16*)(sqp + 3 * (size_t)Bn * Nn);
  __bf16 *fTb[3], *fnb[3], *yb[3], *xb16[3];
  for (int s = 0; s < 3; ++s) {
    fTb[s]  = wsb + (size_t)(0 + s) * fsz;
    fnb[s]  = wsb + (size_t)(3 + s) * fsz;
    yb[s]   = wsb + (size_t)(6 + s) * fsz;
    xb16[s] = wsb + (size_t)(9 + s) * fsz;
  }
  __bf16* Pb = wsb + 12 * fsz;               // [B][n][n] bf16

  zero_kernel<<<(Bn * Cc * Nn + 255) / 256, 256, 0, stream>>>(out, Bn * Cc * Nn);
  for (int s = 0; s < 3; ++s) {
    cast_kernel<<<(Bn * Cc * Nn + 255) / 256, 256, 0, stream>>>(x[s], xb16[s]);
    prep_kernel<<<(Bn * Nn + 255) / 256, 256, 0, stream>>>(x[s], fTb[s], fnb[s], sq[s]);
    conv1x1_kernel<<<(Bn * Nn * Cc + 255) / 256, 256, 0, stream>>>(x[s], w[s], bias[s], yb[s]);
  }

  dim3 gGram(NT, NT, Bn);
  dim3 gRows(Nn, Bn);
  dim3 gGemm(NT, Cc / 16, Bn);
  for (int s = 0; s < 3; ++s)
    for (int t = 0; t < 3; ++t) {
      if (s == t)
        gram_kernel<<<gGram, 32, 0, stream>>>(fTb[s], fTb[t], Dbuf, sq[s], 0);
      else
        gram_kernel<<<gGram, 32, 0, stream>>>(fnb[s], fnb[t], Dbuf, nullptr, 1);
      zero_kernel<<<1, 32, 0, stream>>>(stats, 4);
      stats_kernel<<<dim3(256, Bn), 256, 0, stream>>>(Dbuf, stats);
      adjacency_kernel<<<gRows, 256, 0, stream>>>(Dbuf, stats, Abuf);
      gram_kernel<<<gGram, 32, 0, stream>>>(yb[s], yb[t], Dbuf, nullptr, 2);   // S
      attn_kernel<<<gRows, 256, 0, stream>>>(Abuf, Dbuf, Pb);                  // P (bf16)
      gemm_acc_kernel<<<gGemm, 32, 0, stream>>>(Pb, xb16[t], out);
    }
}
